// RoPESelfAttention_57526791962644
// MI455X (gfx1250) — compile-verified, matrix operand order checked by reading, not run on hardware
//
#include <hip/hip_runtime.h>
#include <stdint.h>

#define B_  4
#define N_  4096
#define D_  1024
#define H_  16
#define HD_ 64

typedef __attribute__((ext_vector_type(16))) __bf16 v16bf;
typedef __attribute__((ext_vector_type(8)))  float  v8f;
typedef __attribute__((ext_vector_type(4)))  float  f32x4;
typedef __attribute__((ext_vector_type(2)))  float  f32x2;
typedef __attribute__((ext_vector_type(4)))  unsigned int u32x4;

static __device__ __forceinline__ unsigned int f2bf1(float f) {
    unsigned int u = __float_as_uint(f);
    return (u + 0x7FFFu + ((u >> 16) & 1u)) >> 16;   // round-to-nearest-even
}
static __device__ __forceinline__ unsigned short f2bfs(float f) {
    return (unsigned short)f2bf1(f);
}
static __device__ __forceinline__ u32x4 pack8(f32x4 a, f32x4 b) {
    u32x4 u;
    u.x = f2bf1(a.x) | (f2bf1(a.y) << 16);
    u.y = f2bf1(a.z) | (f2bf1(a.w) << 16);
    u.z = f2bf1(b.x) | (f2bf1(b.y) << 16);
    u.w = f2bf1(b.z) | (f2bf1(b.w) << 16);
    return u;
}

// Async global->LDS copy of 16 bytes per lane (CDNA5 GLOBAL_LOAD_ASYNC_TO_LDS,
// ASYNCcnt-tracked). LDS destination byte address is the low 32 bits of the
// generic pointer (flat shared aperture keeps the offset in addr[31:0]).
static __device__ __forceinline__ void async_copy_b128(void* lds_dst,
                                                       const void* gsrc) {
    unsigned int loff = (unsigned int)(uintptr_t)lds_dst;
    unsigned long long gaddr = (unsigned long long)(uintptr_t)gsrc;
    asm volatile("global_load_async_to_lds_b128 %0, %1, off"
                 :: "v"(loff), "v"(gaddr) : "memory");
}
static __device__ __forceinline__ void wait_async0() {
    asm volatile("s_wait_asynccnt 0x0" ::: "memory");
}

// 16-bit WMMA operand fragment gather (ISA 7.12.2): lane holds row (lane&15);
// VGPRs 0-3 hold K = (lane<16?0:8)+{0..7} packed pairs, VGPRs 4-7 hold +16.
// Both halves are contiguous in K -> exactly two 16-byte loads per lane.
// B operand assumed symmetric with N on lanes (memory K-contiguous per N row).
static __device__ __forceinline__ v16bf load_frag(const unsigned short* base,
                                                  int row0, int ld, int k0, int lane) {
    const unsigned short* p = base + (size_t)(row0 + (lane & 15)) * ld
                                   + k0 + ((lane >> 4) << 3);
    union { u32x4 u[2]; v16bf v; } f;
    f.u[0] = *(const u32x4*)(p);
    f.u[1] = *(const u32x4*)(p + 16);
    return f.v;
}

static __device__ __forceinline__ v8f wmma_bf16(v16bf a, v16bf b, v8f c) {
    // (neg_a, A, neg_b, B, c_mod, C, reuse_a, reuse_b)
    return __builtin_amdgcn_wmma_f32_16x16x32_bf16(false, a, false, b,
                                                   (short)0, c, false, false);
}

static __device__ __forceinline__ v8f zero8() {
    v8f z = {0.f, 0.f, 0.f, 0.f, 0.f, 0.f, 0.f, 0.f};
    return z;
}

// ---------------------------------------------------------------------------
// Kernel 1: qkv = x @ W_qkv^T, fused RoPE on q,k.
// Q is pre-scaled by HD^-0.5 * log2(e) so attention softmax can use raw
// v_exp_f32 (exp2) without a per-element multiply.
// Writes bf16 Q,K as [B,H,N,64]; V transposed as [B,H,64,N].
// Workgroup tile 64(M) x 128(E); 8 waves in 4x2; wave tile 16x64.
// ---------------------------------------------------------------------------
__global__ __launch_bounds__(256) void qkv_rope_kernel(
    const float* __restrict__ x, const float* __restrict__ rot,
    const float* __restrict__ Wqkv,
    unsigned short* __restrict__ Qb, unsigned short* __restrict__ Kb,
    unsigned short* __restrict__ Vt)
{
    __shared__ unsigned short ldsA[64 * 32];
    __shared__ unsigned short ldsB[128 * 32];
    const int t = threadIdx.x, lane = t & 31, wave = t >> 5;
    const int wr = wave >> 1, wc = wave & 1;
    const int m0 = blockIdx.x * 64;     // token-row block (B*N flat)
    const int e0 = blockIdx.y * 128;    // output-channel block (3D)

    v8f acc[4] = { zero8(), zero8(), zero8(), zero8() };

    for (int k0 = 0; k0 < D_; k0 += 32) {
        {   // A tile: 64 rows x 32 K of x, f32 -> bf16
            int r = t >> 2, c = (t & 3) << 3;
            const float* src = x + (size_t)(m0 + r) * D_ + k0 + c;
            f32x4 a = *(const f32x4*)src;
            f32x4 b = *(const f32x4*)(src + 4);
            *(u32x4*)&ldsA[r * 32 + c] = pack8(a, b);
            if (k0 + 32 < D_) __builtin_prefetch(src + 32, 0, 1);
        }
        {   // B tile: 128 rows (e) x 32 K of W_qkv, f32 -> bf16
            int r = t >> 1, c = (t & 1) << 4;
            const float* src = Wqkv + (size_t)(e0 + r) * D_ + k0 + c;
            f32x4 f0 = *(const f32x4*)(src);
            f32x4 f1 = *(const f32x4*)(src + 4);
            f32x4 f2 = *(const f32x4*)(src + 8);
            f32x4 f3 = *(const f32x4*)(src + 12);
            *(u32x4*)&ldsB[r * 32 + c]     = pack8(f0, f1);
            *(u32x4*)&ldsB[r * 32 + c + 8] = pack8(f2, f3);
            if (k0 + 32 < D_) __builtin_prefetch(src + 32, 0, 1);
        }
        __syncthreads();
        v16bf a = load_frag(ldsA, 16 * wr, 32, 0, lane);
#pragma unroll
        for (int j = 0; j < 4; ++j) {
            v16bf b = load_frag(ldsB, 64 * wc + 16 * j, 32, 0, lane);
            acc[j] = wmma_bf16(a, b, acc[j]);
        }
        __syncthreads();
    }

    // Epilogue: wave tile cols e0+64*wc .. +63 == exactly one (q/k/v, head).
    const int ebase = e0 + 64 * wc;
    const int which = ebase >> 10;             // 0=q, 1=k, 2=v
    const int h     = (ebase & 1023) >> 6;
    const int rowb  = m0 + 16 * wr + ((lane >> 4) << 3);  // C-layout row base
    const float QSCALE = 0.125f * 1.44269504088896340736f;  // HD^-0.5 * log2(e)
#pragma unroll
    for (int j = 0; j < 4; ++j) {
        int hd = 16 * j + (lane & 15);
#pragma unroll
        for (int r = 0; r < 8; ++r) {
            float val = acc[j][r];
            int g = rowb + r;                  // flat token index
            int b = g >> 12;                   // /N_
            int n = g & (N_ - 1);
            if (which < 2) {
                // Rotary pair (hd, hd^1) lives in lane^1 (same row).
                float pair = __shfl_xor(val, 1, 32);
                f32x2 cs = *(const f32x2*)(rot + (((size_t)g * D_) + h * HD_ + hd) * 2);
                float v2 = (hd & 1) ? (val * cs.x + pair * cs.y)
                                    : (val * cs.x - pair * cs.y);
                if (which == 0) v2 *= QSCALE;
                unsigned short* dst = (which == 0) ? Qb : Kb;
                dst[(((size_t)(b * H_ + h)) * N_ + n) * HD_ + hd] = f2bfs(v2);
            } else {
                // V transposed: [B,H,64(hd),N] so PV B-operand is key-contiguous
                Vt[(((size_t)(b * H_ + h)) * HD_ + hd) * (size_t)N_ + n] = f2bfs(val);
            }
        }
    }
}

// ---------------------------------------------------------------------------
// Kernel 2: flash attention. Grid (N/128, B*H); 8 waves, 16 q-rows per wave.
// K/V blocks streamed global->LDS with async-to-LDS (no VGPR round trip).
// Scores arrive pre-scaled by log2(e): softmax uses raw v_exp_f32 (exp2).
// ---------------------------------------------------------------------------
__global__ __launch_bounds__(256) void attn_kernel(
    const unsigned short* __restrict__ Qb, const unsigned short* __restrict__ Kb,
    const unsigned short* __restrict__ Vt, unsigned short* __restrict__ Ob)
{
    __shared__ unsigned short ldsK[64 * 64];       // keys x hd
    __shared__ unsigned short ldsV[64 * 64];       // hd x keys (from Vt)
    __shared__ unsigned short ldsP[8 * 16 * 64];   // per-wave P staging
    const int t = threadIdx.x, lane = t & 31, wave = t >> 5;
    const int bh = blockIdx.y;
    const int q0 = blockIdx.x * 128 + wave * 16;
    const size_t qkBase = (size_t)bh * N_ * HD_;
    const size_t vtBase = (size_t)bh * HD_ * N_;

    // Resident Q A-fragments (hd chunks 0-31, 32-63); Q already scaled.
    v16bf qf[2];
    {
        const unsigned short* qrow = Qb + qkBase
            + (size_t)(q0 + (lane & 15)) * HD_ + ((lane >> 4) << 3);
#pragma unroll
        for (int c = 0; c < 2; ++c) {
            union { u32x4 u[2]; v16bf v; } f;
            f.u[0] = *(const u32x4*)(qrow + 32 * c);
            f.u[1] = *(const u32x4*)(qrow + 32 * c + 16);
            qf[c] = f.v;
        }
    }

    float mrow[8], lrow[8];
    v8f o[4] = { zero8(), zero8(), zero8(), zero8() };
#pragma unroll
    for (int r = 0; r < 8; ++r) { mrow[r] = -1e30f; lrow[r] = 0.f; }

    unsigned short* pw = ldsP + wave * (16 * 64);

    for (int kb = 0; kb < N_; kb += 64) {
        {   // async stream: 64-key K block and 64x64 Vt block, 64B per thread
            int r = t >> 2, c = (t & 3) << 4;
            const unsigned short* sk = Kb + qkBase + (size_t)(kb + r) * HD_ + c;
            async_copy_b128(&ldsK[r * 64 + c],     sk);
            async_copy_b128(&ldsK[r * 64 + c + 8], sk + 8);
            const unsigned short* sv = Vt + vtBase + (size_t)r * N_ + kb + c;
            async_copy_b128(&ldsV[r * 64 + c],     sv);
            async_copy_b128(&ldsV[r * 64 + c + 8], sv + 8);
        }
        wait_async0();
        __syncthreads();

        // S = Q @ K^T over hd (two 32-chunks), 4 key sub-blocks of 16
        v8f s[4];
#pragma unroll
        for (int j = 0; j < 4; ++j) {
            s[j] = zero8();
            s[j] = wmma_bf16(qf[0], load_frag(ldsK, 16 * j, 64, 0,  lane), s[j]);
            s[j] = wmma_bf16(qf[1], load_frag(ldsK, 16 * j, 64, 32, lane), s[j]);
        }

        // Online softmax in exp2 domain; row spans 16 lanes of each half-wave.
#pragma unroll
        for (int r = 0; r < 8; ++r) {
            float mc = fmaxf(fmaxf(s[0][r], s[1][r]), fmaxf(s[2][r], s[3][r]));
            mc = fmaxf(mc, __shfl_xor(mc, 1, 32));
            mc = fmaxf(mc, __shfl_xor(mc, 2, 32));
            mc = fmaxf(mc, __shfl_xor(mc, 4, 32));
            mc = fmaxf(mc, __shfl_xor(mc, 8, 32));
            float mn = fmaxf(mrow[r], mc);
            float sc = __builtin_amdgcn_exp2f(mrow[r] - mn);
            float p0 = __builtin_amdgcn_exp2f(s[0][r] - mn);
            float p1 = __builtin_amdgcn_exp2f(s[1][r] - mn);
            float p2 = __builtin_amdgcn_exp2f(s[2][r] - mn);
            float p3 = __builtin_amdgcn_exp2f(s[3][r] - mn);
            s[0][r] = p0; s[1][r] = p1; s[2][r] = p2; s[3][r] = p3;
            float rs = p0 + p1 + p2 + p3;
            rs += __shfl_xor(rs, 1, 32);
            rs += __shfl_xor(rs, 2, 32);
            rs += __shfl_xor(rs, 4, 32);
            rs += __shfl_xor(rs, 8, 32);
            lrow[r] = lrow[r] * sc + rs;
            mrow[r] = mn;
#pragma unroll
            for (int j = 0; j < 4; ++j) o[j][r] *= sc;
        }

        // Stage P bf16 to per-wave LDS (C layout -> row-major), reload as A frags
        {
            int rbase = (lane >> 4) << 3, cidx = lane & 15;
#pragma unroll
            for (int j = 0; j < 4; ++j)
#pragma unroll
                for (int r = 0; r < 8; ++r)
                    pw[(size_t)(rbase + r) * 64 + 16 * j + cidx] = f2bfs(s[j][r]);
        }
        __syncthreads();
        v16bf pf0 = load_frag(pw, 0, 64, 0,  lane);
        v16bf pf1 = load_frag(pw, 0, 64, 32, lane);
        // O += P @ V : contraction over keys (two 32-chunks), 4 hd sub-blocks
#pragma unroll
        for (int j = 0; j < 4; ++j) {
            o[j] = wmma_bf16(pf0, load_frag(ldsV, 16 * j, 64, 0,  lane), o[j]);
            o[j] = wmma_bf16(pf1, load_frag(ldsV, 16 * j, 64, 32, lane), o[j]);
        }
        __syncthreads();
    }

    // Normalize and store O as bf16 in [B*N, D] layout for the final GEMM.
    const int b = bh >> 4, h = bh & 15;
#pragma unroll
    for (int r = 0; r < 8; ++r) {
        float inv = 1.f / lrow[r];
        int n = q0 + r + ((lane >> 4) << 3);
        size_t rowBase = ((size_t)(b * N_ + n)) * D_ + h * HD_;
#pragma unroll
        for (int j = 0; j < 4; ++j)
            Ob[rowBase + 16 * j + (lane & 15)] = f2bfs(o[j][r] * inv);
    }
}

// ---------------------------------------------------------------------------
// Kernel 3: out = O @ W_out^T + b_out (f32 output). A tile is a pure bf16
// copy -> streamed with async-to-LDS; W_out needs f32->bf16 via VGPRs.
// ---------------------------------------------------------------------------
__global__ __launch_bounds__(256) void out_proj_kernel(
    const unsigned short* __restrict__ Ob, const float* __restrict__ Wout,
    const float* __restrict__ bout, float* __restrict__ out)
{
    __shared__ unsigned short ldsA[64 * 32];
    __shared__ unsigned short ldsB[128 * 32];
    const int t = threadIdx.x, lane = t & 31, wave = t >> 5;
    const int wr = wave >> 1, wc = wave & 1;
    const int m0 = blockIdx.x * 64;
    const int e0 = blockIdx.y * 128;

    v8f acc[4] = { zero8(), zero8(), zero8(), zero8() };

    for (int k0 = 0; k0 < D_; k0 += 32) {
        {   // A tile: already bf16, async copy 16B per thread
            int r = t >> 2, c = (t & 3) << 3;
            async_copy_b128(&ldsA[r * 32 + c],
                            Ob + (size_t)(m0 + r) * D_ + k0 + c);
        }
        {   // B tile: W_out rows, f32 -> bf16
            int r = t >> 1, c = (t & 1) << 4;
            const float* src = Wout + (size_t)(e0 + r) * D_ + k0 + c;
            f32x4 f0 = *(const f32x4*)(src);
            f32x4 f1 = *(const f32x4*)(src + 4);
            f32x4 f2 = *(const f32x4*)(src + 8);
            f32x4 f3 = *(const f32x4*)(src + 12);
            *(u32x4*)&ldsB[r * 32 + c]     = pack8(f0, f1);
            *(u32x4*)&ldsB[r * 32 + c + 8] = pack8(f2, f3);
            if (k0 + 32 < D_) __builtin_prefetch(src + 32, 0, 1);
        }
        wait_async0();
        __syncthreads();
        v16bf a = load_frag(ldsA, 16 * wr, 32, 0, lane);
#pragma unroll
        for (int j = 0; j < 4; ++j)
            acc[j] = wmma_bf16(a, load_frag(ldsB, 64 * wc + 16 * j, 32, 0, lane), acc[j]);
        __syncthreads();
    }

    const int rowb = m0 + 16 * wr + ((lane >> 4) << 3);
#pragma unroll
    for (int j = 0; j < 4; ++j) {
        int e = e0 + 64 * wc + 16 * j + (lane & 15);
        float bj = bout[e];
#pragma unroll
        for (int r = 0; r < 8; ++r)
            out[(size_t)(rowb + r) * D_ + e] = acc[j][r] + bj;
    }
}

// ---------------------------------------------------------------------------
extern "C" void kernel_launch(void* const* d_in, const int* in_sizes, int n_in,
                              void* d_out, int out_size, void* d_ws, size_t ws_size,
                              hipStream_t stream) {
    const float* x    = (const float*)d_in[0];   // [B,N,D]
    const float* rot  = (const float*)d_in[1];   // [B,N,D,2]
    const float* Wqkv = (const float*)d_in[2];   // [3D,D]
    const float* Wout = (const float*)d_in[3];   // [D,D]
    const float* bout = (const float*)d_in[4];   // [D]
    float* out = (float*)d_out;

    // Workspace: 4 bf16 segments of B*H*N*64 = 16,777,216 elems (33.5 MB each)
    unsigned short* ws = (unsigned short*)d_ws;
    const size_t SEG = (size_t)B_ * H_ * N_ * HD_;
    unsigned short* Qb = ws;
    unsigned short* Kb = ws + SEG;
    unsigned short* Vt = ws + 2 * SEG;
    unsigned short* Ob = ws + 3 * SEG;

    qkv_rope_kernel<<<dim3((B_ * N_) / 64, (3 * D_) / 128), 256, 0, stream>>>(
        x, rot, Wqkv, Qb, Kb, Vt);
    attn_kernel<<<dim3(N_ / 128, B_ * H_), 256, 0, stream>>>(Qb, Kb, Vt, Ob);
    out_proj_kernel<<<dim3((B_ * N_) / 64, D_ / 128), 256, 0, stream>>>(
        Ob, Wout, bout, out);

    (void)in_sizes; (void)n_in; (void)out_size; (void)ws_size;
}
